// LSTMClassifier_5403068858755
// MI455X (gfx1250) — compile-verified, matrix operand order checked by reading, not run on hardware
//
#include <hip/hip_runtime.h>

typedef __attribute__((ext_vector_type(16))) _Float16 v16h;
typedef __attribute__((ext_vector_type(8)))  _Float16 v8h;
typedef __attribute__((ext_vector_type(8)))  float    v8f;

#define EMB   10
#define HID   16
#define TLEN  512

// CDNA5 native V_TANH_F32 (confirmed in emitted asm); exp-based fallback kept for portability.
#if __has_builtin(__builtin_amdgcn_tanhf)
__device__ __forceinline__ float fast_tanh(float x) { return __builtin_amdgcn_tanhf(x); }
#else
__device__ __forceinline__ float fast_tanh(float x) {
    float e = __expf(2.0f * x);
    return 1.0f - 2.0f * __builtin_amdgcn_rcpf(e + 1.0f);
}
#endif
// sigmoid(x) where the *pre-activation was already scaled by 0.5* (folded into weights):
//   sig(x) = 0.5*tanh(0.5x)+0.5  ->  p=0.5x baked in  ->  fma(0.5, tanh(p), 0.5)
__device__ __forceinline__ float sig_prescaled(float p) { return fmaf(0.5f, fast_tanh(p), 0.5f); }

// Transposed-product formulation:  gates^T = Wcomb @ [xe|h]^T  per 16-gate tile.
//   A (16x32 f16) = 16 rows of Wcomb for gate tile g (scaled), fused K-space:
//     K=0..9  -> Wih dims, K=10..15 -> zero, K=16..31 -> Whh dims 0..15
//   16-bit A layout: lane holds M=lane&15 (gate row); element e<8 -> K = hi*8+e,
//                    element e>=8 -> K = 16 + hi*8 + (e-8).
__device__ __forceinline__ v16h buildA(const float* __restrict__ Wih,
                                       const float* __restrict__ Whh,
                                       int g, int m, int hi, float scale) {
    v16h a;
#pragma unroll
    for (int e = 0; e < 16; ++e) {
        float v;
        if (e < 8) {                                  // xe half of K-space
            int K = hi * 8 + e;
            v = (K < EMB) ? Wih[(g * 16 + m) * EMB + K] : 0.0f;
        } else {                                      // h half of K-space
            int j = hi * 8 + (e - 8);                 // hidden index 0..15
            v = Whh[(g * 16 + m) * HID + j];
        }
        a[e] = (_Float16)(v * scale);
    }
    return a;
}

// One LSTM timestep: 4 independent v_wmma_f32_16x16x32_f16 + lane-local cell.
// New h packs straight into B[8..15] (v_cvt_pk into the WMMA operand registers).
__device__ __forceinline__ void lstm_step(const v16h (&wA)[4], const v8f (&biasC)[4],
                                          v16h& B, v8f& c8, v8f& h8) {
    v8f p[4];
#pragma unroll
    for (int g = 0; g < 4; ++g)
        p[g] = __builtin_amdgcn_wmma_f32_16x16x32_f16(
                   false, wA[g], false, B, (short)0, biasC[g], false, false);
#pragma unroll
    for (int r = 0; r < 8; ++r) {
        float iv = sig_prescaled(p[0][r]);   // i/f/o tiles arrive pre-scaled by 0.5
        float fv = sig_prescaled(p[1][r]);
        float gv = fast_tanh   (p[2][r]);
        float ov = sig_prescaled(p[3][r]);
        float cv = fmaf(fv, c8[r], iv * gv);
        c8[r] = cv;
        float hval = ov * fast_tanh(cv);
        h8[r] = hval;
        B[8 + r] = (_Float16)hval;
    }
}

__global__ __launch_bounds__(256) void lstm_fused_kernel(
    const int*   __restrict__ x,    // [B, T]
    const float* __restrict__ emb,  // [VOCAB+1, EMB]
    const float* __restrict__ Wih,  // [64, EMB]
    const float* __restrict__ Whh,  // [64, HID]
    const float* __restrict__ bih,  // [64]
    const float* __restrict__ bhh,  // [64]
    const float* __restrict__ Wfc,  // [1, HID]
    const float* __restrict__ bfc,  // [1]
    float*       __restrict__ out)  // [B]
{
    const int tid  = threadIdx.x;
    const int w    = tid >> 5;
    const int lane = tid & 31;
    const int n    = lane & 15;   // batch row within tile (B/D column) AND gate row for A build
    const int hi   = lane >> 4;   // half-wave selector
    const int rowBase = (blockIdx.x * 8 + w) * 16;

    // ---- weight A-fragments + bias-as-C fragments (built once, live in VGPRs)
    // gate tiles: 0=i, 1=f, 2=g, 3=o ; sigmoid tiles pre-scaled by 0.5
    v16h wA[4];
    v8f  biasC[4];
#pragma unroll
    for (int g = 0; g < 4; ++g) {
        const float sc = (g == 2) ? 1.0f : 0.5f;
        wA[g] = buildA(Wih, Whh, g, n, hi, sc);
        // D layout: VGPR r holds gate component m = hi*8 + r (same for every lane/row)
#pragma unroll
        for (int r = 0; r < 8; ++r) {
            int gi = g * 16 + hi * 8 + r;
            biasC[g][r] = (bih[gi] + bhh[gi]) * sc;
        }
    }
    float wfc8[8];
#pragma unroll
    for (int r = 0; r < 8; ++r) wfc8[r] = Wfc[hi * 8 + r];
    const float bfc0 = bfc[0];

    // ---- state: lane holds c/h[row=n][hid = hi*8 + r]
    v8f c8, h8;
#pragma unroll
    for (int r = 0; r < 8; ++r) { c8[r] = 0.0f; h8[r] = 0.0f; }

    const int* xp = x + (rowBase + n) * TLEN;

    // ---- persistent B fragment ([xe|h]^T, 32x16 f16), loop-carried in registers:
    //      elements 0..7 = packed xe (rotated each step), 8..15 = packed h (written by cell)
    // lanes 0-15 carry emb dims 0..7; lanes 16-31 carry dims 8,9 then zeros
    // (zeros double as the K=10..15 padding of the fused B fragment)
    v16h B;
    {
        int idx0 = xp[0];
        const float* er = emb + idx0 * EMB;
        if (hi == 0) {
#pragma unroll
            for (int k = 0; k < 8; ++k) B[k] = (_Float16)er[k];
        } else {
            B[0] = (_Float16)er[8]; B[1] = (_Float16)er[9];
#pragma unroll
            for (int k = 2; k < 8; ++k) B[k] = (_Float16)0.0f;
        }
#pragma unroll
        for (int e = 8; e < 16; ++e) B[e] = (_Float16)0.0f;   // h(t=-1) = 0
    }

    // ---- main loop: T-1 steps with a 1-deep gather pipeline, last step peeled.
    // xp[t+1] is always in-bounds here; the prefetch is *speculative* per the ISA
    // (translation faults silently dropped), so no bounds guard and the address
    // strength-reduces to a pointer increment.
#pragma unroll 1
    for (int t = 0; t < TLEN - 1; ++t) {
        float en[8];
        {
            int idxn = xp[t + 1];
            const float* ern = emb + idxn * EMB;
            if (hi == 0) {
#pragma unroll
                for (int k = 0; k < 8; ++k) en[k] = ern[k];
            } else {
                en[0] = ern[8]; en[1] = ern[9];
#pragma unroll
                for (int k = 2; k < 8; ++k) en[k] = 0.0f;
            }
        }
        __builtin_prefetch(xp + t + 64, 0, 1);  // global_prefetch_b8, speculative

        lstm_step(wA, biasC, B, c8, h8);

        // rotate the embedding pipeline into B's lower half (off the critical path)
#pragma unroll
        for (int k = 0; k < 8; ++k) B[k] = (_Float16)en[k];
    }
    // ---- peeled final step: no gather, no rotation
    lstm_step(wA, biasC, B, c8, h8);

    // ---- head: out[row] = sigmoid(h . Wfc + bfc)
    // lane pair (L, L+16) holds complementary hid halves of row n: one xor-shuffle
    float s = 0.0f;
#pragma unroll
    for (int r = 0; r < 8; ++r) s = fmaf(h8[r], wfc8[r], s);
    s += __shfl_xor(s, 16, 32);
    if (hi == 0)
        out[rowBase + n] = fmaf(0.5f, fast_tanh(0.5f * (s + bfc0)), 0.5f);  // 16 contiguous floats
}

extern "C" void kernel_launch(void* const* d_in, const int* in_sizes, int n_in,
                              void* d_out, int out_size, void* d_ws, size_t ws_size,
                              hipStream_t stream) {
    const int*   x   = (const int*)  d_in[0];
    const float* emb = (const float*)d_in[1];
    const float* Wih = (const float*)d_in[2];
    const float* Whh = (const float*)d_in[3];
    const float* bih = (const float*)d_in[4];
    const float* bhh = (const float*)d_in[5];
    const float* Wfc = (const float*)d_in[6];
    const float* bfc = (const float*)d_in[7];
    float* out = (float*)d_out;

    const int B = in_sizes[0] / TLEN;         // 4096
    const int blocks = B / (16 * 8);          // 16 rows/wave * 8 waves/block = 128 rows/block
    lstm_fused_kernel<<<dim3(blocks), dim3(256), 0, stream>>>(
        x, emb, Wih, Whh, bih, bhh, Wfc, bfc, out);
}
